// TorchTransformer_5720896438290
// MI455X (gfx1250) — matrix operand order repaired and checked by reading, not compiled, not run on hardware
//
#include <hip/hip_runtime.h>
#include <hip/hip_bf16.h>
#include <float.h>
#include <math.h>
#include <stdint.h>

// ---------------------------------------------------------------------------
// Model constants (reference: V,S,L,H,D,DF = 32000,2048,8,16,1024,4096)
// ---------------------------------------------------------------------------
constexpr int Bc = 2, Sc = 2048, Lc = 8, Hc = 16, Dc = 1024, DFc = 4096, Vc = 32000;
constexpr int DHc = Dc / Hc;            // 64
constexpr int Mc  = Bc * Sc;            // 4096 rows through every GEMM
constexpr float EPSc = 1.1920928955078125e-07f;  // jnp.finfo(f32).eps

typedef __bf16 bf16_t;
typedef __bf16 v16bf __attribute__((ext_vector_type(16)));
typedef __bf16 v8bf  __attribute__((ext_vector_type(8)));
typedef float  v8f   __attribute__((ext_vector_type(8)));
typedef unsigned int u32x4 __attribute__((ext_vector_type(4)));
typedef int          i32x4 __attribute__((ext_vector_type(4)));
typedef int          i32x8 __attribute__((ext_vector_type(8)));

// CDNA5 WMMA: D(f32 16x16) = A(bf16 16x32) * B(bf16 32x16) + C
__device__ __forceinline__ v8f wmma_bf16(v16bf a, v16bf b, v8f c) {
  return __builtin_amdgcn_wmma_f32_16x16x32_bf16(
      /*neg_a=*/false, a, /*neg_b=*/false, b,
      /*c_mod=*/(short)0, c, /*reuse_a=*/false, /*reuse_b=*/false);
}

// A-fragment (16x32 bf16) lane layout: lane L holds row (L&15),
// contiguous K {half*16 .. half*16+15}, half = L>>4 (matches B fragments).
__device__ __forceinline__ v16bf load_a_frag(const bf16_t* row_base, int hf) {
  v8bf lo = *(const v8bf*)(row_base + hf * 16);
  v8bf hi = *(const v8bf*)(row_base + hf * 16 + 8);
  return __builtin_shufflevector(lo, hi, 0, 1, 2, 3, 4, 5, 6, 7,
                                         8, 9, 10, 11, 12, 13, 14, 15);
}

// ---------------------------------------------------------------------------
// Tensor Data Mover: stage one 128-row x 32-elem bf16 tile (row-major source,
// row stride = strideElems) into LDS at lds_addr. LDS padding: after every
// 16 DWORDs (one 32-elem row) insert 8 DWORDs -> row stride 48 elems (96B),
// matching the compute-side LDT and keeping 32B alignment for fragments.
// D# layout per CDNA5 ISA.
// ---------------------------------------------------------------------------
__device__ __forceinline__ void tdm_load_tile_128x32(unsigned lds_addr,
                                                     const void* gptr,
                                                     unsigned strideElems,
                                                     unsigned tensorRows) {
  const unsigned long long ga = (unsigned long long)(uintptr_t)gptr;
  // ---- group 0 (128b): count=1 | lds_addr | global_addr | type=2 ----
  const u32x4 g0 = {1u, lds_addr, (unsigned)(ga & 0xFFFFFFFFu),
                    (unsigned)((ga >> 32) & 0x01FFFFFFu) | (2u << 30)};
  // ---- group 1 (256b) packed via 64-bit words ----
  const unsigned long long dim0   = strideElems;  // tensor width (elems)
  const unsigned long long dim1   = tensorRows;   // tensor height (rows)
  const unsigned long long str0   = strideElems;  // dim0 stride (elems)
  const unsigned long long tileW  = 32, tileH = 128;
  unsigned long long w0 = (1ull << 16)      // data_size = 2B
                        | (1ull << 20)      // pad_enable
                        | (3ull << 22)      // pad_interval: 16 DWORDs
                        | (7ull << 25);     // pad_amount:   8 DWORDs
  w0 |= (dim0 & 0xFFFFull) << 48;                       // tensor_dim0[15:0]
  unsigned long long w1 = ((dim0 >> 16) & 0xFFFFull)    // tensor_dim0[31:16]
                        | ((dim1 & 0xFFFFFFFFull) << 16)// tensor_dim1
                        | (tileW << 48);                // tile_dim0
  unsigned long long w2 = tileH                         // tile_dim1
                        | ((str0 & 0xFFFFFFFFull) << 32);  // stride[31:0]
  unsigned long long w3 = (str0 >> 32) & 0xFFFFull;     // stride[47:32]
  const i32x8 g1 = {(int)w0, (int)(w0 >> 32), (int)w1, (int)(w1 >> 32),
                    (int)w2, (int)(w2 >> 32), (int)w3, (int)(w3 >> 32)};
  const i32x4 z4 = {0, 0, 0, 0};
#if defined(__clang_major__) && __clang_major__ >= 23
  const i32x8 z8 = {0, 0, 0, 0, 0, 0, 0, 0};
  __builtin_amdgcn_tensor_load_to_lds(g0, g1, z4, z4, z8, 0);
#else
  __builtin_amdgcn_tensor_load_to_lds(g0, g1, z4, z4, 0);
#endif
}

// ---------------------------------------------------------------------------
// GEMM: C[M,N](f32 or bf16) = A[M,K](bf16) * B[N,K](bf16, pre-transposed)
// Block = 256 thr = 8 waves; tile 128x128, K-step 32; wave tile 32x64 (8 WMMA).
// Tiles are streamed by the Tensor Data Mover (wave 0 issues, TENSORcnt
// pipelines two tiles in flight, double-buffered LDS).
// ---------------------------------------------------------------------------
template <bool OUTBF>
__global__ __launch_bounds__(256)
void gemm_bf16_nt_kernel(const bf16_t* __restrict__ A,
                         const bf16_t* __restrict__ Bm,
                         void* __restrict__ Cout, int M, int N, int K) {
  constexpr int LDT = 48;  // row stride produced by TDM padding (96B, 32B-aligned)
  __shared__ bf16_t lda[2][128 * LDT];
  __shared__ bf16_t ldb[2][128 * LDT];

  const int tid  = threadIdx.x;
  const int lane = tid & 31;
  const int wid  = tid >> 5;
  const int wm   = wid >> 1;           // 0..3  -> 32-row slice
  const int wn   = wid & 1;            // 0..1  -> 64-col slice
  const int n0   = blockIdx.x * 128;
  const int m0   = blockIdx.y * 128;
  const int hf   = lane >> 4;
  const int lr   = lane & 15;
  const int nt   = K / 32;

  v8f acc[2][4];
  const v8f vzero = {0.f, 0.f, 0.f, 0.f, 0.f, 0.f, 0.f, 0.f};
#pragma unroll
  for (int i = 0; i < 2; ++i)
#pragma unroll
    for (int j = 0; j < 4; ++j) acc[i][j] = vzero;

  auto issue = [&](int t) {
    const int buf = t & 1;
    tdm_load_tile_128x32((unsigned)(uintptr_t)&lda[buf][0],
                         &A[(size_t)m0 * K + t * 32], (unsigned)K, (unsigned)M);
    tdm_load_tile_128x32((unsigned)(uintptr_t)&ldb[buf][0],
                         &Bm[(size_t)n0 * K + t * 32], (unsigned)K, (unsigned)N);
  };

  if (wid == 0) issue(0);

  for (int t = 0; t < nt; ++t) {
    if (wid == 0) {
      if (t + 1 < nt) {
        issue(t + 1);                          // overlap next tile's DMA
        __builtin_amdgcn_s_wait_tensorcnt(2);  // tile t resident (in-order)
      } else {
        __builtin_amdgcn_s_wait_tensorcnt(0);
      }
    }
    __syncthreads();

    const bf16_t* la = lda[t & 1];
    const bf16_t* lb = ldb[t & 1];
    v16bf af[2], bfg[4];
#pragma unroll
    for (int i = 0; i < 2; ++i)
      af[i] = load_a_frag(&la[(wm * 32 + i * 16 + lr) * LDT], hf);
#pragma unroll
    for (int j = 0; j < 4; ++j)  // B lane layout: col = lr, 16 contiguous K per half
      bfg[j] = *(const v16bf*)&lb[(wn * 64 + j * 16 + lr) * LDT + hf * 16];
#pragma unroll
    for (int i = 0; i < 2; ++i)
#pragma unroll
      for (int j = 0; j < 4; ++j) acc[i][j] = wmma_bf16(af[i], bfg[j], acc[i][j]);

    __syncthreads();  // all waves done reading before buf (t&1) is re-filled
  }

  // C layout: VGPR r -> row_local r + 8*hf, col_local = lr.
#pragma unroll
  for (int i = 0; i < 2; ++i)
#pragma unroll
    for (int j = 0; j < 4; ++j)
#pragma unroll
      for (int r = 0; r < 8; ++r) {
        const size_t row = (size_t)(m0 + wm * 32 + i * 16 + r + 8 * hf);
        const int    col = n0 + wn * 64 + j * 16 + lr;
        const float  v   = acc[i][j][r];
        if (OUTBF) ((bf16_t*)Cout)[row * N + col] = (bf16_t)v;
        else       ((float*)Cout)[row * N + col]  = v;
      }
}

// ---------------------------------------------------------------------------
// Flash attention, causal. Grid (B*H, S/128), block 256 = 8 waves.
// Each wave owns one 16-row Q tile; KV streamed in 32-key blocks; scores never
// touch HBM. softmax stats per row via half-wave (width-16) shuffles.
// ---------------------------------------------------------------------------
__global__ __launch_bounds__(256)
void flash_attn_kernel(const bf16_t* __restrict__ Q, const bf16_t* __restrict__ Kb,
                       const bf16_t* __restrict__ Vb, bf16_t* __restrict__ O) {
  constexpr int LDT = 48;
  __shared__ bf16_t lds_p[8][16 * LDT];   // per-wave P patch (16q x 32kv)
  __shared__ bf16_t lds_v[8][64 * LDT];   // per-wave V^T patch (64dh x 32kv)

  const int tid  = threadIdx.x;
  const int lane = tid & 31;
  const int wid  = tid >> 5;
  const int b    = blockIdx.x / Hc;
  const int h    = blockIdx.x % Hc;
  const int q0   = (blockIdx.y * 8 + wid) * 16;
  const size_t mb = (size_t)b * Sc;
  const int hf = lane >> 4, lr = lane & 15;
  const int hcol = h * DHc;

  // Q A-fragments for K-chunks 0..31 and 32..63 of the 64-wide head dim.
  v16bf aq[2];
#pragma unroll
  for (int kc = 0; kc < 2; ++kc) {
    const bf16_t* qp = &Q[(mb + q0 + lr) * Dc + hcol + kc * 32];
    v8bf lo = *(const v8bf*)(qp + hf * 16);
    v8bf hi = *(const v8bf*)(qp + hf * 16 + 8);
    aq[kc] = __builtin_shufflevector(lo, hi, 0, 1, 2, 3, 4, 5, 6, 7,
                                             8, 9, 10, 11, 12, 13, 14, 15);
  }

  const v8f vzero = {0.f, 0.f, 0.f, 0.f, 0.f, 0.f, 0.f, 0.f};
  v8f o[4] = {vzero, vzero, vzero, vzero};
  float ms[8], ls[8];
#pragma unroll
  for (int r = 0; r < 8; ++r) { ms[r] = -1e30f; ls[r] = 0.f; }

  bf16_t* pp = lds_p[wid];
  bf16_t* vp = lds_v[wid];

  for (int kvb = 0; kvb < q0 + 16; kvb += 32) {
    if (kvb + 32 < q0 + 16) {  // hint next KV block (global_prefetch_b8)
      __builtin_prefetch(&Kb[(mb + kvb + 32 + lane) * Dc + hcol], 0, 1);
      __builtin_prefetch(&Vb[(mb + kvb + 32 + lane) * Dc + hcol], 0, 1);
    }
    // ---- S = Q * K^T for two 16-key sub-tiles ----
    v8f s[2] = {vzero, vzero};
#pragma unroll
    for (int n = 0; n < 2; ++n)
#pragma unroll
      for (int kc = 0; kc < 2; ++kc) {
        const v16bf bk = *(const v16bf*)&Kb[(mb + kvb + n * 16 + lr) * Dc +
                                            hcol + kc * 32 + hf * 16];
        s[n] = wmma_bf16(aq[kc], bk, s[n]);
      }

    // ---- scale + causal mask ----
#pragma unroll
    for (int n = 0; n < 2; ++n) {
      const int kj = kvb + n * 16 + lr;
#pragma unroll
      for (int r = 0; r < 8; ++r) {
        const int qi = q0 + r + 8 * hf;
        const float v = s[n][r] * 0.125f;        // 1/sqrt(DH=64)
        s[n][r] = (kj <= qi) ? v : -1e30f;
      }
    }

    // ---- online softmax (rows live across 16 lanes of a half-wave) ----
#pragma unroll
    for (int r = 0; r < 8; ++r) {
      float vmax = fmaxf(s[0][r], s[1][r]);
#pragma unroll
      for (int off = 1; off < 16; off <<= 1) vmax = fmaxf(vmax, __shfl_xor(vmax, off, 16));
      const float mnew  = fmaxf(ms[r], vmax);
      const float alpha = __expf(ms[r] - mnew);
      const float p0 = __expf(s[0][r] - mnew);
      const float p1 = __expf(s[1][r] - mnew);
      float rs = p0 + p1;
#pragma unroll
      for (int off = 1; off < 16; off <<= 1) rs += __shfl_xor(rs, off, 16);
      ls[r] = ls[r] * alpha + rs;
      ms[r] = mnew;
#pragma unroll
      for (int t = 0; t < 4; ++t) o[t][r] *= alpha;
      s[0][r] = p0; s[1][r] = p1;
    }

    // ---- P (C-layout) -> per-wave LDS, re-read in A-fragment order ----
#pragma unroll
    for (int r = 0; r < 8; ++r) {
      const int prow = r + 8 * hf;
      pp[prow * LDT + lr]      = (bf16_t)s[0][r];
      pp[prow * LDT + 16 + lr] = (bf16_t)s[1][r];
    }

    // ---- stage V block transposed: lane = kv row, scatter into [dh][kv] ----
    {
      const bf16_t* vg = &Vb[(mb + kvb + lane) * Dc + hcol];
#pragma unroll
      for (int c = 0; c < 4; ++c) {
        const v16bf vv = *(const v16bf*)(vg + c * 16);
#pragma unroll
        for (int e = 0; e < 16; ++e) vp[(c * 16 + e) * LDT + lane] = vv[e];
      }
    }

    // ---- O += P * V ----
    const v16bf ap = load_a_frag(&pp[lr * LDT], hf);
#pragma unroll
    for (int t = 0; t < 4; ++t) {
      const v16bf bv = *(const v16bf*)&vp[(t * 16 + lr) * LDT + hf * 16];
      o[t] = wmma_bf16(ap, bv, o[t]);
    }
  }

  // ---- normalize and emit bf16 head slice ----
#pragma unroll
  for (int t = 0; t < 4; ++t)
#pragma unroll
    for (int r = 0; r < 8; ++r) {
      const size_t row = mb + q0 + r + 8 * hf;
      const int    col = hcol + t * 16 + lr;
      O[row * Dc + col] = (bf16_t)(o[t][r] / ls[r]);
    }
}

// ---------------------------------------------------------------------------
// Elementwise / reduction kernels
// ---------------------------------------------------------------------------
__global__ __launch_bounds__(256)
void embed_kernel(const int* __restrict__ ids, const float* __restrict__ TE,
                  const float* __restrict__ PE, float* __restrict__ X) {
  const int m = blockIdx.x, t = threadIdx.x;
  const int s = m % Sc;
  const int id = ids[m];
  const float4 a = ((const float4*)(TE + (size_t)id * Dc))[t];
  const float4 p = ((const float4*)(PE + (size_t)s * Dc))[t];
  ((float4*)(X + (size_t)m * Dc))[t] =
      make_float4(a.x + p.x, a.y + p.y, a.z + p.z, a.w + p.w);
}

__global__ __launch_bounds__(256)
void rmsnorm_bf16_kernel(const float* __restrict__ X, const float* __restrict__ W,
                         bf16_t* __restrict__ Y) {
  __shared__ float red[8];
  const int m = blockIdx.x, tid = threadIdx.x;
  const float4 v = ((const float4*)(X + (size_t)m * Dc))[tid];
  float ss = v.x * v.x + v.y * v.y + v.z * v.z + v.w * v.w;
#pragma unroll
  for (int off = 16; off > 0; off >>= 1) ss += __shfl_xor(ss, off, 32);
  if ((tid & 31) == 0) red[tid >> 5] = ss;
  __syncthreads();
  float tot = 0.f;
#pragma unroll
  for (int i = 0; i < 8; ++i) tot += red[i];
  const float inv = rsqrtf(tot * (1.0f / (float)Dc) + EPSc);
  const float4 w = ((const float4*)W)[tid];
  bf16_t* y = Y + (size_t)m * Dc + tid * 4;
  y[0] = (bf16_t)(v.x * inv * w.x);
  y[1] = (bf16_t)(v.y * inv * w.y);
  y[2] = (bf16_t)(v.z * inv * w.z);
  y[3] = (bf16_t)(v.w * inv * w.w);
}

__global__ __launch_bounds__(256)
void add_inplace_kernel(float* __restrict__ X, const float* __restrict__ Y, int n4) {
  const int i = blockIdx.x * 256 + threadIdx.x;
  if (i < n4) {
    float4 a = ((float4*)X)[i];
    const float4 b = ((const float4*)Y)[i];
    ((float4*)X)[i] = make_float4(a.x + b.x, a.y + b.y, a.z + b.z, a.w + b.w);
  }
}

__global__ __launch_bounds__(256)
void gelu_bf16_kernel(const float* __restrict__ X, bf16_t* __restrict__ Y, int n) {
  const int i = blockIdx.x * 256 + threadIdx.x;
  if (i < n) {
    const float v = X[i];
    Y[i] = (bf16_t)(0.5f * v * (1.0f + erff(v * 0.70710678118654752f)));
  }
}

__global__ __launch_bounds__(256)
void cvt_bf16_kernel(const float* __restrict__ X, bf16_t* __restrict__ Y, int n) {
  const int i = blockIdx.x * 256 + threadIdx.x;
  if (i < n) Y[i] = (bf16_t)X[i];
}

// Transpose-convert weight: in[K,N] f32 -> out[N,K] bf16 (tiled, LDS 32x33)
__global__ __launch_bounds__(256)
void cvtT_bf16_kernel(const float* __restrict__ X, bf16_t* __restrict__ Y,
                      int K, int N) {
  __shared__ float tile[32][33];
  const int nb = blockIdx.x * 32, kb = blockIdx.y * 32;
  const int tx = threadIdx.x & 31, ty = threadIdx.x >> 5;  // 32 x 8
#pragma unroll
  for (int r = 0; r < 32; r += 8)
    tile[ty + r][tx] = X[(size_t)(kb + ty + r) * N + nb + tx];
  __syncthreads();
#pragma unroll
  for (int r = 0; r < 32; r += 8)
    Y[(size_t)(nb + ty + r) * K + kb + tx] = (bf16_t)tile[tx][ty + r];
}

// ---------------------------------------------------------------------------
// Orchestration
// ---------------------------------------------------------------------------
extern "C" void kernel_launch(void* const* d_in, const int* in_sizes, int n_in,
                              void* d_out, int out_size, void* d_ws, size_t ws_size,
                              hipStream_t stream) {
  (void)in_sizes; (void)n_in; (void)out_size; (void)ws_size;
  const int*   ids = (const int*)d_in[0];
  const float* TE  = (const float*)d_in[1];
  const float* PE  = (const float*)d_in[2];
  const float* WQ  = (const float*)d_in[3];
  const float* WK  = (const float*)d_in[4];
  const float* WV  = (const float*)d_in[5];
  const float* WO  = (const float*)d_in[6];
  const float* W1  = (const float*)d_in[7];
  const float* W2  = (const float*)d_in[8];
  const float* ANW = (const float*)d_in[9];
  const float* FNW = (const float*)d_in[10];
  const float* FLN = (const float*)d_in[11];

  char* ws = (char*)d_ws;
  size_t off = 0;
  auto alloc = [&](size_t bytes) -> void* {
    void* p = ws + off;
    off += (bytes + 255) & ~(size_t)255;
    return p;
  };
  float*  x     = (float*)alloc((size_t)Mc * Dc * 4);
  bf16_t* hbf   = (bf16_t*)alloc((size_t)Mc * Dc * 2);
  bf16_t* qbf   = (bf16_t*)alloc((size_t)Mc * Dc * 2);
  bf16_t* kbf   = (bf16_t*)alloc((size_t)Mc * Dc * 2);
  bf16_t* vbf   = (bf16_t*)alloc((size_t)Mc * Dc * 2);
  bf16_t* abf   = (bf16_t*)alloc((size_t)Mc * Dc * 2);
  float*  gout  = (float*)alloc((size_t)Mc * DFc * 4);
  bf16_t* midbf = (bf16_t*)alloc((size_t)Mc * DFc * 2);
  bf16_t* wT    = (bf16_t*)alloc((size_t)Dc * DFc * 2);
  bf16_t* tebf  = (bf16_t*)alloc((size_t)Vc * Dc * 2);

  const dim3 blk(256);
  const dim3 gDD(Dc / 128, Mc / 128);    // GEMM M x D
  const dim3 gDF(DFc / 128, Mc / 128);   // GEMM M x DF
  const dim3 gLV(Vc / 128, Mc / 128);    // logits GEMM (250 x 32)
  const dim3 gTdd(Dc / 32, Dc / 32);
  const dim3 gTdf(DFc / 32, Dc / 32);    // [D,DF] -> [DF,D]
  const dim3 gTfd(Dc / 32, DFc / 32);    // [DF,D] -> [D,DF]
  const dim3 gFlash(Bc * Hc, Sc / 128);
  const int  n4d = Mc * Dc / 4;

  embed_kernel<<<Mc, blk, 0, stream>>>(ids, TE, PE, x);

  for (int l = 0; l < Lc; ++l) {
    const float* wq = WQ + (size_t)l * Dc * Dc;
    const float* wk = WK + (size_t)l * Dc * Dc;
    const float* wv = WV + (size_t)l * Dc * Dc;
    const float* wo = WO + (size_t)l * Dc * Dc;
    const float* w1 = W1 + (size_t)l * Dc * DFc;
    const float* w2 = W2 + (size_t)l * DFc * Dc;

    // ---- attention ----
    rmsnorm_bf16_kernel<<<Mc, blk, 0, stream>>>(x, ANW + (size_t)l * Dc, hbf);

    cvtT_bf16_kernel<<<gTdd, blk, 0, stream>>>(wq, wT, Dc, Dc);
    gemm_bf16_nt_kernel<true><<<gDD, blk, 0, stream>>>(hbf, wT, qbf, Mc, Dc, Dc);
    cvtT_bf16_kernel<<<gTdd, blk, 0, stream>>>(wk, wT, Dc, Dc);
    gemm_bf16_nt_kernel<true><<<gDD, blk, 0, stream>>>(hbf, wT, kbf, Mc, Dc, Dc);
    cvtT_bf16_kernel<<<gTdd, blk, 0, stream>>>(wv, wT, Dc, Dc);
    gemm_bf16_nt_kernel<true><<<gDD, blk, 0, stream>>>(hbf, wT, vbf, Mc, Dc, Dc);

    flash_attn_kernel<<<gFlash, blk, 0, stream>>>(qbf, kbf, vbf, abf);

    cvtT_bf16_kernel<<<gTdd, blk, 0, stream>>>(wo, wT, Dc, Dc);
    gemm_bf16_nt_kernel<false><<<gDD, blk, 0, stream>>>(abf, wT, gout, Mc, Dc, Dc);
    add_inplace_kernel<<<(n4d + 255) / 256, blk, 0, stream>>>(x, gout, n4d);

    // ---- FFN ----
    rmsnorm_bf16_kernel<<<Mc, blk, 0, stream>>>(x, FNW + (size_t)l * Dc, hbf);
    cvtT_bf16_kernel<<<gTdf, blk, 0, stream>>>(w1, wT, Dc, DFc);
    gemm_bf16_nt_kernel<false><<<gDF, blk, 0, stream>>>(hbf, wT, gout, Mc, DFc, Dc);
    gelu_bf16_kernel<<<(Mc * DFc + 255) / 256, blk, 0, stream>>>(gout, midbf, Mc * DFc);
    cvtT_bf16_kernel<<<gTfd, blk, 0, stream>>>(w2, wT, DFc, Dc);
    gemm_bf16_nt_kernel<false><<<gDD, blk, 0, stream>>>(midbf, wT, gout, Mc, Dc, DFc);
    add_inplace_kernel<<<(n4d + 255) / 256, blk, 0, stream>>>(x, gout, n4d);
  }

  // ---- tied output projection ----
  rmsnorm_bf16_kernel<<<Mc, blk, 0, stream>>>(x, FLN, hbf);
  cvt_bf16_kernel<<<(Vc * Dc + 255) / 256, blk, 0, stream>>>(TE, tebf, Vc * Dc);
  gemm_bf16_nt_kernel<false><<<gLV, blk, 0, stream>>>(hbf, tebf, (float*)d_out,
                                                      Mc, Vc, Dc);
}
